// Attention_4449586119407
// MI455X (gfx1250) — compile-verified, matrix operand order checked by reading, not run on hardware
//
#include <hip/hip_runtime.h>

#define BATCH 2
#define SEQ   2048
#define DIMSZ 2048
#define NH    32
#define NKV   8
#define HD    64
#define SCALE 0.125f   // HEAD_DIM^-0.5 = 1/8

typedef __bf16 bf16;
typedef __bf16 v16bf __attribute__((ext_vector_type(16)));
typedef float  v8f   __attribute__((ext_vector_type(8)));

union Frag { v16bf v; float4 f4[2]; };

// Fragment loader for both A (rows, K-contiguous) and B (cols of B=W^T = rows
// of W, K-contiguous). 16-bit WMMA layout (ISA 7.12.2):
//   elements 0..7  = K[kBase + hi*8 + 0..7]
//   elements 8..15 = K[kBase + 16 + hi*8 + 0..7]
__device__ inline v16bf load_frag(const bf16* __restrict__ base, int row, int ld,
                                  int kBase, int lane) {
    int hi = (lane >> 4) & 1;
    const bf16* p = base + (size_t)row * ld + kBase + hi * 8;
    Frag fr;
    fr.f4[0] = *(const float4*)(p);
    fr.f4[1] = *(const float4*)(p + 16);
    return fr.v;
}

__device__ inline v8f zero8() {
    v8f z;
    #pragma unroll
    for (int i = 0; i < 8; ++i) z[i] = 0.0f;
    return z;
}

// ---------------- f32 -> bf16 cast, 8 elems / thread, 16B stores ----------------
__global__ void cast_kernel(const float* __restrict__ in, bf16* __restrict__ out, size_t n) {
    size_t i = ((size_t)blockIdx.x * blockDim.x + threadIdx.x) * 8;
    if (i >= n) return;
    float4 a = *(const float4*)(in + i);
    float4 b = *(const float4*)(in + i + 4);
    union { bf16 h[8]; float4 v; } u;
    u.h[0] = (bf16)a.x; u.h[1] = (bf16)a.y; u.h[2] = (bf16)a.z; u.h[3] = (bf16)a.w;
    u.h[4] = (bf16)b.x; u.h[5] = (bf16)b.y; u.h[6] = (bf16)b.z; u.h[7] = (bf16)b.w;
    *(float4*)(out + i) = u.v;
}

// ---------------- GEMM: C[M,N] = A[M,K] * W[N,K]^T ----------------
// block = 128 (4 waves); each wave computes a 32x64 tile (2 A-frags reuse 4 B-frags).
// grid = (M/32, N/256)
template <bool F32OUT>
__global__ __launch_bounds__(128)
void gemm_rt_kernel(const bf16* __restrict__ A, const bf16* __restrict__ W,
                    void* __restrict__ Cout, int M, int N, int K) {
    int lane  = threadIdx.x & 31;
    int wave  = threadIdx.x >> 5;
    int hi    = lane >> 4;
    int tileM = blockIdx.x * 32;
    int nBase = blockIdx.y * 256 + wave * 64;
    if (nBase >= N) return;                 // wave-uniform
    int r0 = tileM + (lane & 15);
    int r1 = r0 + 16;

    v8f acc[2][4];
    #pragma unroll
    for (int i = 0; i < 2; ++i)
        #pragma unroll
        for (int t = 0; t < 4; ++t) acc[i][t] = zero8();

    for (int kb = 0; kb < K; kb += 32) {
        if (kb + 256 < K) {  // uniform; lowers to global_prefetch_b8
            __builtin_prefetch(A + (size_t)r0 * K + kb + 256, 0, 1);
            __builtin_prefetch(W + (size_t)(nBase + (lane & 15)) * K + kb + 256, 0, 1);
        }
        v16bf a0 = load_frag(A, r0, K, kb, lane);
        v16bf a1 = load_frag(A, r1, K, kb, lane);
        #pragma unroll
        for (int t = 0; t < 4; ++t) {
            v16bf b = load_frag(W, nBase + t * 16 + (lane & 15), K, kb, lane);
            acc[0][t] = __builtin_amdgcn_wmma_f32_16x16x32_bf16(
                false, a0, false, b, (short)0, acc[0][t], false, false);
            acc[1][t] = __builtin_amdgcn_wmma_f32_16x16x32_bf16(
                false, a1, false, b, (short)0, acc[1][t], false, false);
        }
    }

    #pragma unroll
    for (int i = 0; i < 2; ++i)
        #pragma unroll
        for (int t = 0; t < 4; ++t)
            #pragma unroll
            for (int r = 0; r < 8; ++r) {
                int row = tileM + i * 16 + r + hi * 8;
                int col = nBase + t * 16 + (lane & 15);
                float val = acc[i][t][r];
                if (F32OUT) ((float*)Cout)[(size_t)row * N + col] = val;
                else        ((bf16*)Cout)[(size_t)row * N + col] = (bf16)val;
            }
}

// ---------------- RoPE + reshape [B,L,H,64] -> [B,H,L,64] ----------------
__global__ void rope_kernel(const bf16* __restrict__ proj, const float* __restrict__ cosT,
                            const float* __restrict__ sinT, bf16* __restrict__ out,
                            int nHeads) {
    size_t idx = (size_t)blockIdx.x * blockDim.x + threadIdx.x; // [B,L,nHeads,32]
    size_t total = (size_t)BATCH * SEQ * nHeads * 32;
    if (idx >= total) return;
    int    i  = (int)(idx & 31);
    size_t t1 = idx >> 5;
    int    h  = (int)(t1 % nHeads);
    size_t t2 = t1 / nHeads;
    int    l  = (int)(t2 % SEQ);
    int    b  = (int)(t2 / SEQ);
    size_t src = ((size_t)(b * SEQ + l) * nHeads + h) * HD + 2 * i;
    float te = (float)proj[src];
    float to = (float)proj[src + 1];
    float c = cosT[l * (HD / 2) + i];
    float s = sinT[l * (HD / 2) + i];
    size_t dst = ((size_t)(b * nHeads + h) * SEQ + l) * HD + 2 * i;
    out[dst]     = (bf16)(te * c - to * s);
    out[dst + 1] = (bf16)(te * s + to * c);
}

// ---------------- V transpose: [B,L,KVh,64] -> [B,KVh,64,L] ----------------
__global__ void vtrans_kernel(const bf16* __restrict__ vproj, bf16* __restrict__ vt) {
    size_t idx = (size_t)blockIdx.x * blockDim.x + threadIdx.x; // [B,L,NKV,HD]
    size_t total = (size_t)BATCH * SEQ * NKV * HD;
    if (idx >= total) return;
    int    d  = (int)(idx & (HD - 1));
    size_t t1 = idx >> 6;
    int    h  = (int)(t1 % NKV);
    size_t t2 = t1 / NKV;
    int    l  = (int)(t2 % SEQ);
    int    b  = (int)(t2 / SEQ);
    vt[((size_t)(b * NKV + h) * HD + d) * SEQ + l] = vproj[idx];
}

// ---------------- Flash attention: one wave per 32-query tile ----------------
// grid = (B*NH, SEQ/32), block = 32.
// 2 query row-tiles share every K/V fragment: 16 WMMAs per 8KB of K/V traffic.
__global__ __launch_bounds__(32)
void attn_kernel(const bf16* __restrict__ q, const bf16* __restrict__ k,
                 const bf16* __restrict__ vt, bf16* __restrict__ out) {
    __shared__ bf16 ldsP[32 * 32];   // P tile: 32 queries x 32 keys, bf16
    int lane = threadIdx.x & 31;
    int hi   = lane >> 4;
    int bh   = blockIdx.x;
    int b    = bh / NH;
    int h    = bh % NH;
    int kvh  = h / (NH / NKV);
    int qBase = blockIdx.y * 32;

    const bf16* qHead = q  + (size_t)bh * SEQ * HD;
    const bf16* kHead = k  + (size_t)(b * NKV + kvh) * SEQ * HD;
    const bf16* vHead = vt + (size_t)(b * NKV + kvh) * HD * SEQ;

    // Q fragments: [qtile][kstep]
    v16bf qf[2][2];
    #pragma unroll
    for (int i = 0; i < 2; ++i) {
        int rowQ = qBase + i * 16 + (lane & 15);
        qf[i][0] = load_frag(qHead, rowQ, HD, 0, lane);
        qf[i][1] = load_frag(qHead, rowQ, HD, 32, lane);
    }

    v8f acc[2][4];
    #pragma unroll
    for (int i = 0; i < 2; ++i)
        #pragma unroll
        for (int t = 0; t < 4; ++t) acc[i][t] = zero8();
    float m[2][8], lsum[2][8];
    #pragma unroll
    for (int i = 0; i < 2; ++i)
        #pragma unroll
        for (int r = 0; r < 8; ++r) { m[i][r] = -1e30f; lsum[i][r] = 0.0f; }

    for (int kb = 0; kb < qBase + 32; kb += 32) {
        // S = Q (32x64) * K_block^T (64x32): sf[qtile][ktile]
        v8f sf[2][2];
        #pragma unroll
        for (int t = 0; t < 2; ++t) {
            int krow = kb + t * 16 + (lane & 15);
            v16bf kf0 = load_frag(kHead, krow, HD, 0, lane);
            v16bf kf1 = load_frag(kHead, krow, HD, 32, lane);
            #pragma unroll
            for (int i = 0; i < 2; ++i) {
                v8f c = zero8();
                c = __builtin_amdgcn_wmma_f32_16x16x32_bf16(false, qf[i][0], false, kf0,
                                                            (short)0, c, false, false);
                c = __builtin_amdgcn_wmma_f32_16x16x32_bf16(false, qf[i][1], false, kf1,
                                                            (short)0, c, false, false);
                sf[i][t] = c;
            }
        }

        // scale + causal mask, per-row online softmax (rows split across 16-lane halves)
        float s[2][2][8], mnew[2][8];
        #pragma unroll
        for (int i = 0; i < 2; ++i)
            #pragma unroll
            for (int r = 0; r < 8; ++r) {
                int qrow = qBase + i * 16 + r + hi * 8;
                mnew[i][r] = m[i][r];
                #pragma unroll
                for (int t = 0; t < 2; ++t) {
                    int key = kb + t * 16 + (lane & 15);
                    float v = sf[i][t][r] * SCALE;
                    if (key > qrow) v = -1e9f;
                    s[i][t][r] = v;
                    mnew[i][r] = fmaxf(mnew[i][r], v);
                }
            }
        #pragma unroll
        for (int i = 0; i < 2; ++i)
            #pragma unroll
            for (int r = 0; r < 8; ++r)
                #pragma unroll
                for (int off = 1; off < 16; off <<= 1)
                    mnew[i][r] = fmaxf(mnew[i][r], __shfl_xor(mnew[i][r], off, 16));

        float fac[2][8], psum[2][8];
        #pragma unroll
        for (int i = 0; i < 2; ++i)
            #pragma unroll
            for (int r = 0; r < 8; ++r) {
                fac[i][r] = __expf(m[i][r] - mnew[i][r]);
                float p0 = __expf(s[i][0][r] - mnew[i][r]);
                float p1 = __expf(s[i][1][r] - mnew[i][r]);
                s[i][0][r] = p0; s[i][1][r] = p1;
                psum[i][r] = p0 + p1;
            }
        #pragma unroll
        for (int i = 0; i < 2; ++i)
            #pragma unroll
            for (int r = 0; r < 8; ++r)
                #pragma unroll
                for (int off = 1; off < 16; off <<= 1)
                    psum[i][r] += __shfl_xor(psum[i][r], off, 16);
        #pragma unroll
        for (int i = 0; i < 2; ++i)
            #pragma unroll
            for (int r = 0; r < 8; ++r) {
                lsum[i][r] = lsum[i][r] * fac[i][r] + psum[i][r];
                m[i][r] = mnew[i][r];
            }
        #pragma unroll
        for (int i = 0; i < 2; ++i)
            #pragma unroll
            for (int t = 0; t < 4; ++t)
                #pragma unroll
                for (int r = 0; r < 8; ++r)
                    acc[i][t][r] = acc[i][t][r] * fac[i][r];

        // P (C-layout f32) -> LDS bf16 32x32 -> A-fragment layout
        __syncthreads();
        #pragma unroll
        for (int i = 0; i < 2; ++i)
            #pragma unroll
            for (int t = 0; t < 2; ++t)
                #pragma unroll
                for (int r = 0; r < 8; ++r)
                    ldsP[(i * 16 + r + hi * 8) * 32 + t * 16 + (lane & 15)] =
                        (bf16)s[i][t][r];
        __syncthreads();
        v16bf pa[2];
        pa[0] = load_frag(ldsP, (lane & 15), 32, 0, lane);
        pa[1] = load_frag(ldsP, 16 + (lane & 15), 32, 0, lane);

        // O += P (32x32) * V_block (32x64): V^T rows are key-contiguous
        #pragma unroll
        for (int t = 0; t < 4; ++t) {
            v16bf vb = load_frag(vHead, t * 16 + (lane & 15), SEQ, kb, lane);
            #pragma unroll
            for (int i = 0; i < 2; ++i)
                acc[i][t] = __builtin_amdgcn_wmma_f32_16x16x32_bf16(
                    false, pa[i], false, vb, (short)0, acc[i][t], false, false);
        }
    }

    // write attn output [B*L, NH*HD] bf16 (input to final projection)
    #pragma unroll
    for (int i = 0; i < 2; ++i)
        #pragma unroll
        for (int t = 0; t < 4; ++t)
            #pragma unroll
            for (int r = 0; r < 8; ++r) {
                int row = b * SEQ + qBase + i * 16 + r + hi * 8;
                int col = h * HD + t * 16 + (lane & 15);
                out[(size_t)row * (NH * HD) + col] = (bf16)(acc[i][t][r] / lsum[i][r]);
            }
}

extern "C" void kernel_launch(void* const* d_in, const int* in_sizes, int n_in,
                              void* d_out, int out_size, void* d_ws, size_t ws_size,
                              hipStream_t stream) {
    const float* x    = (const float*)d_in[0];
    const float* wq   = (const float*)d_in[1];
    const float* wk   = (const float*)d_in[2];
    const float* wv   = (const float*)d_in[3];
    const float* wo   = (const float*)d_in[4];
    const float* fcos = (const float*)d_in[5];
    const float* fsin = (const float*)d_in[6];

    const size_t nX  = (size_t)BATCH * SEQ * DIMSZ;       // 8388608
    const size_t nWq = (size_t)DIMSZ * DIMSZ;             // 4194304
    const size_t nWk = (size_t)(NKV * HD) * DIMSZ;        // 1048576
    const size_t nKV = (size_t)BATCH * SEQ * NKV * HD;    // 2097152
    const int M = BATCH * SEQ;                            // 4096

    char* ws = (char*)d_ws;
    size_t off = 0;
    auto alloc = [&](size_t bytes) -> char* {
        char* p = ws + off;
        off += (bytes + 255) & ~(size_t)255;
        return p;
    };
    bf16* xb    = (bf16*)alloc(nX  * 2);
    bf16* wqb   = (bf16*)alloc(nWq * 2);
    bf16* wkb   = (bf16*)alloc(nWk * 2);
    bf16* wvb   = (bf16*)alloc(nWk * 2);
    bf16* wob   = (bf16*)alloc(nWq * 2);
    bf16* qproj = (bf16*)alloc(nX  * 2);
    bf16* kproj = (bf16*)alloc(nKV * 2);
    bf16* vproj = (bf16*)alloc(nKV * 2);
    bf16* qr    = (bf16*)alloc(nX  * 2);
    bf16* kr    = (bf16*)alloc(nKV * 2);
    bf16* vt    = (bf16*)alloc(nKV * 2);
    bf16* attn  = (bf16*)alloc(nX  * 2);

    auto blocks8 = [](size_t n) { return (unsigned)((n / 8 + 255) / 256); };
    auto blocks  = [](size_t n) { return (unsigned)((n + 255) / 256); };

    // 1) casts to bf16 (vectorized)
    cast_kernel<<<blocks8(nX),  256, 0, stream>>>(x,  xb,  nX);
    cast_kernel<<<blocks8(nWq), 256, 0, stream>>>(wq, wqb, nWq);
    cast_kernel<<<blocks8(nWk), 256, 0, stream>>>(wk, wkb, nWk);
    cast_kernel<<<blocks8(nWk), 256, 0, stream>>>(wv, wvb, nWk);
    cast_kernel<<<blocks8(nWq), 256, 0, stream>>>(wo, wob, nWq);

    // 2) QKV projections
    gemm_rt_kernel<false><<<dim3(M / 32, DIMSZ / 256), 128, 0, stream>>>(
        xb, wqb, qproj, M, DIMSZ, DIMSZ);
    gemm_rt_kernel<false><<<dim3(M / 32, (NKV * HD) / 256), 128, 0, stream>>>(
        xb, wkb, kproj, M, NKV * HD, DIMSZ);
    gemm_rt_kernel<false><<<dim3(M / 32, (NKV * HD) / 256), 128, 0, stream>>>(
        xb, wvb, vproj, M, NKV * HD, DIMSZ);

    // 3) RoPE (q,k) + V transpose
    rope_kernel<<<blocks((size_t)BATCH * SEQ * NH * 32), 256, 0, stream>>>(
        qproj, fcos, fsin, qr, NH);
    rope_kernel<<<blocks((size_t)BATCH * SEQ * NKV * 32), 256, 0, stream>>>(
        kproj, fcos, fsin, kr, NKV);
    vtrans_kernel<<<blocks(nKV), 256, 0, stream>>>(vproj, vt);

    // 4) causal flash attention (32-query tiles)
    attn_kernel<<<dim3(BATCH * NH, SEQ / 32), 32, 0, stream>>>(qr, kr, vt, attn);

    // 5) output projection -> f32 d_out
    gemm_rt_kernel<true><<<dim3(M / 32, DIMSZ / 256), 128, 0, stream>>>(
        attn, wob, (float*)d_out, M, DIMSZ, DIMSZ);
}
